// ConceptFlow_52252572123549
// MI455X (gfx1250) — matrix operand order repaired and checked by reading, not compiled, not run on hardware
//
#include <hip/hip_runtime.h>
#include <math.h>

// ---------------------------------------------------------------------------
// ConceptFlow forward for MI455X (gfx1250, wave32, WMMA).
// v2: all WMMA operands are f16 with K padded to a multiple of 32 so fragment
// loads are unconditional aligned vector loads (no per-element exec juggling).
// fp32 master copies + f16 shadows (fused into GEMM epilogues) since every
// big tensor feeds 2-3 consumers -> halves read bytes on a bandwidth-bound
// workload (~45 GFLOP vs ~1 GB traffic @ 23.3 TB/s).
// ---------------------------------------------------------------------------

typedef __attribute__((ext_vector_type(16))) _Float16 v16h;
typedef __attribute__((ext_vector_type(8)))  _Float16 v8h;
typedef __attribute__((ext_vector_type(8)))  float    v8f;

#define GF_RELU   1
#define GF_TRANS  2
#define GF_STOREH 4

namespace cf {
constexpr int B = 64, E = 1000, F = 4000, LQ = 30, T = 100, EMB = 300, K3 = 300;
constexpr float VERY_NEG = -100000000000.0f;
constexpr float EPSV     = 1e-10f;
constexpr float LAMBDA   = 0.8f;
}

__device__ __forceinline__ float sigf(float x) { return 1.0f / (1.0f + expf(-x)); }

// A-fragment (16x32 f16): lane row = m0+(lane&15); halfs[0..15] =
// K kbase..kbase+15, kbase = ks*32+(lane>>4)*16 (same order as B).
// One aligned 32B load.
__device__ __forceinline__ v16h load_a16(const _Float16* __restrict__ p) {
  return *(const v16h*)p;
}

// ---------------------------------------------------------------------------
// Y = act(alpha * X(gather) @ W^T + bias*rowscale) ; optional transposed fp32
// store and f16 shadow store (zero-padded to ldyh). X, W are f16 with padded
// leading dims (lda, ldb multiples of 32 halfs -> 64B aligned rows).
// Block = 256 threads = 8 waves: 4 M-subtiles x 2 N-subtiles of 16x16.
// ---------------------------------------------------------------------------
__global__ void wmma_gemm(const _Float16* __restrict__ X, const _Float16* __restrict__ W,
                          const float* __restrict__ bias,
                          const float* __restrict__ bias_rowscale,
                          const int* __restrict__ rowidx,
                          float* __restrict__ Y, _Float16* __restrict__ Yh,
                          int M, int N, int Kp, int lda, int ldb, int ldy, int ldyh,
                          long long xbs, long long wbs, long long ybs, long long ybsh,
                          float alpha, int flags) {
  const int b = blockIdx.z;
  const _Float16* Xb = X + xbs * (long long)b;
  const _Float16* Wb = W + wbs * (long long)b;
  float* Yb = Y + ybs * (long long)b;
  _Float16* Yhb = Yh ? Yh + ybsh * (long long)b : (_Float16*)0;

  const int lane = threadIdx.x & 31;
  const int wave = threadIdx.x >> 5;
  const int m0 = blockIdx.x * 64 + (wave & 3) * 16;
  const int n0 = blockIdx.y * 32 + (wave >> 2) * 16;

  int am = m0 + (lane & 15); if (am > M - 1) am = M - 1;
  long long arow = rowidx ? (long long)rowidx[am] : (long long)am;
  int bn = n0 + (lane & 15); if (bn > N - 1) bn = N - 1;

  const _Float16* ap = Xb + arow * (long long)lda + ((lane >> 4) << 4);
  const _Float16* bp = Wb + (long long)bn * ldb + ((lane >> 4) << 4);

  v8f acc = {};
  const int nk = Kp >> 5;
  for (int ks = 0; ks < nk; ++ks) {
    v16h a = load_a16(ap + ks * 32);
    v16h bf = *(const v16h*)(bp + ks * 32);
    acc = __builtin_amdgcn_wmma_f32_16x16x32_f16(false, a, false, bf, (short)0, acc,
                                                 false, false);
  }

  const int hi = (lane >> 4) << 3;
  const int gn = n0 + (lane & 15);
  float vals[8];
#pragma unroll
  for (int v = 0; v < 8; ++v) {
    float val = acc[v] * alpha;
    if (bias) val += bias[gn] * (bias_rowscale ? bias_rowscale[m0 + v + hi] : 1.0f);
    if (flags & GF_RELU) val = fmaxf(val, 0.0f);
    vals[v] = val;
  }

  if (m0 + 16 <= M && n0 + 16 <= N) {  // interior fast path: unguarded stores
    if (flags & GF_TRANS) {
#pragma unroll
      for (int v = 0; v < 8; ++v) Yb[(long long)gn * ldy + (m0 + v + hi)] = vals[v];
    } else {
#pragma unroll
      for (int v = 0; v < 8; ++v) Yb[(long long)(m0 + v + hi) * ldy + gn] = vals[v];
      if (flags & GF_STOREH) {
#pragma unroll
        for (int v = 0; v < 8; ++v)
          Yhb[(long long)(m0 + v + hi) * ldyh + gn] = (_Float16)vals[v];
      }
    }
  } else {  // edge tiles: f16 shadow writes zeros in [N, ldyh) pad columns
#pragma unroll
    for (int v = 0; v < 8; ++v) {
      int gm = m0 + v + hi;
      if (gm < M) {
        if (gn < N) {
          if (flags & GF_TRANS) Yb[(long long)gn * ldy + gm] = vals[v];
          else                  Yb[(long long)gm * ldy + gn] = vals[v];
        }
        if ((flags & GF_STOREH) && gn < ldyh)
          Yhb[(long long)gm * ldyh + gn] = (gn < N) ? (_Float16)vals[v] : (_Float16)0.0f;
      }
    }
  }
}

// ---------------------------------------------------------------------------
// LSTM recurrence. xw = x@Wih^T + (bih+bhh) precomputed. h kept f16 in LDS
// (64x128, zero padded); 25 waves own one 16-col tile of the 64x400 gates.
// Emits padded-f16 Q (B,LQ,128), Q^T (B,T,32) and final hidden (B,128).
// ---------------------------------------------------------------------------
__global__ void lstm_kernel(const float* __restrict__ xw, const _Float16* __restrict__ Whh16,
                            _Float16* __restrict__ Qh, _Float16* __restrict__ QT,
                            _Float16* __restrict__ qnode16) {
  extern __shared__ char smem_raw[];
  _Float16* h16 = (_Float16*)smem_raw;                                // 64*128 f16
  float* g = (float*)(smem_raw + 64 * 128 * 2);                       // 64*400 f32
  float* c = (float*)(smem_raw + 64 * 128 * 2 + 64 * 400 * 4);        // 64*100 f32

  const int tid = threadIdx.x;
  const int lane = tid & 31;
  const int wave = tid >> 5;

  for (int i = tid; i < 64 * 128; i += (int)blockDim.x) h16[i] = (_Float16)0.0f;
  for (int i = tid; i < 64 * 100; i += (int)blockDim.x) c[i] = 0.0f;
  // zero f16 outputs (pad columns stay zero)
  for (int i = tid; i < 64 * cf::LQ * 128; i += (int)blockDim.x) Qh[i] = (_Float16)0.0f;
  for (int i = tid; i < 64 * cf::T * 32; i += (int)blockDim.x) QT[i] = (_Float16)0.0f;
  for (int i = tid; i < 64 * 128; i += (int)blockDim.x) qnode16[i] = (_Float16)0.0f;
  __syncthreads();

  const int ka_off = (lane >> 4) << 4;
  const int kb_off = (lane >> 4) << 4;
  const int hi = (lane >> 4) << 3;

  for (int t = 0; t < cf::LQ; ++t) {
    if (wave < 25) {
      const int n0 = wave * 16;
      const int gn = n0 + (lane & 15);
      const _Float16* bp = Whh16 + (long long)gn * 128 + kb_off;
      for (int msub = 0; msub < 4; ++msub) {
        const int m0 = msub * 16;
        const int am = m0 + (lane & 15);
        v8f acc;
#pragma unroll
        for (int v = 0; v < 8; ++v)
          acc[v] = xw[((long long)(m0 + v + hi) * cf::LQ + t) * 400 + gn];
        const _Float16* apl = h16 + am * 128 + ka_off;
#pragma unroll
        for (int ks = 0; ks < 4; ++ks) {
          v16h a = load_a16(apl + ks * 32);            // two ds_load_b128
          v16h bf = *(const v16h*)(bp + ks * 32);
          acc = __builtin_amdgcn_wmma_f32_16x16x32_f16(false, a, false, bf, (short)0,
                                                       acc, false, false);
        }
#pragma unroll
        for (int v = 0; v < 8; ++v) g[(m0 + v + hi) * 400 + gn] = acc[v];
      }
    }
    __syncthreads();
    for (int idx = tid; idx < 64 * 100; idx += (int)blockDim.x) {
      int bb = idx / 100, j = idx - bb * 100;
      float gi = g[bb * 400 + j];
      float gf = g[bb * 400 + 100 + j];
      float gg = g[bb * 400 + 200 + j];
      float go = g[bb * 400 + 300 + j];
      float cv = sigf(gf) * c[idx] + sigf(gi) * tanhf(gg);
      float hv = sigf(go) * tanhf(cv);
      c[idx] = cv;
      _Float16 hh = (_Float16)hv;
      h16[bb * 128 + j] = hh;
      Qh[((long long)bb * cf::LQ + t) * 128 + j] = hh;
      QT[((long long)bb * cf::T + j) * 32 + t] = hh;
      if (t == cf::LQ - 1) qnode16[bb * 128 + j] = hh;
    }
    __syncthreads();
  }
}

// ------------------------- helpers -----------------------------------------
__global__ void vec_add_k(const float* a, const float* b, float* o, int n) {
  int i = blockIdx.x * blockDim.x + threadIdx.x;
  if (i < n) o[i] = a[i] + b[i];
}
__global__ void fill_k(float* p, float v, long long n) {
  long long i = (long long)blockIdx.x * blockDim.x + threadIdx.x;
  if (i < n) p[i] = v;
}
// fp32 (rows,K,ldsrc) -> f16 (rows,ldp) zero-padded
__global__ void cvt_pad_k(const float* __restrict__ s, _Float16* __restrict__ d,
                          long long rows, int K, int ldsrc, int ldp) {
  long long i = (long long)blockIdx.x * blockDim.x + threadIdx.x;
  if (i >= rows * (long long)ldp) return;
  long long r = i / ldp;
  int k = (int)(i - r * ldp);
  d[i] = (k < K) ? (_Float16)s[r * ldsrc + k] : (_Float16)0.0f;
}
// gathered rows: d[r,:] = f16(src[idx[r],:K]) zero-padded
__global__ void gather_cvt_k(const float* __restrict__ s, const int* __restrict__ idx,
                             _Float16* __restrict__ d, long long rows, int K, int ldsrc,
                             int ldp) {
  long long i = (long long)blockIdx.x * blockDim.x + threadIdx.x;
  if (i >= rows * (long long)ldp) return;
  long long r = i / ldp;
  int k = (int)(i - r * ldp);
  d[i] = (k < K) ? (_Float16)s[(long long)idx[r] * ldsrc + k] : (_Float16)0.0f;
}
// masked softmax over L=30 for each (b,f); writes f16 row (32 cols, pad=0)
__global__ void softmax_l_k(const float* __restrict__ simTf, _Float16* __restrict__ simT16,
                            const int* __restrict__ qtext, int BF, int Fdim) {
  int idx = blockIdx.x * blockDim.x + threadIdx.x;
  if (idx >= BF) return;
  int b = idx / Fdim;
  const float* row = simTf + (long long)idx * 32;
  _Float16* rowh = simT16 + (long long)idx * 32;
  float v[cf::LQ];
  float mx = -3.4e38f;
#pragma unroll
  for (int l = 0; l < cf::LQ; ++l) {
    float x = row[l] + ((qtext[b * cf::LQ + l] != 0) ? 0.0f : cf::VERY_NEG);
    v[l] = x;
    mx = fmaxf(mx, x);
  }
  float s = 0.0f;
#pragma unroll
  for (int l = 0; l < cf::LQ; ++l) { v[l] = expf(v[l] - mx); s += v[l]; }
  float inv = 1.0f / s;
#pragma unroll
  for (int l = 0; l < cf::LQ; ++l) rowh[l] = (_Float16)(v[l] * inv);
  rowh[30] = (_Float16)0.0f;
  rowh[31] = (_Float16)0.0f;
}
// o[row] = scale * dot(A[row,:L], Bm[row,:L])  (one wave per row)
__global__ void rowdot_k(const float* __restrict__ A, const float* __restrict__ Bm,
                         float* __restrict__ o, long long rows, int L, float scale) {
  long long row = (long long)blockIdx.x * (blockDim.x >> 5) + (threadIdx.x >> 5);
  int lane = threadIdx.x & 31;
  if (row >= rows) return;
  const float* a = A + row * L;
  const float* b = Bm + row * L;
  float s = 0.0f;
  for (int j = lane; j < L; j += 32) s += a[j] * b[j];
  for (int off = 16; off > 0; off >>= 1) s += __shfl_down(s, off, 32);
  if (lane == 0) o[row] = s * scale;
}
__global__ void rowmax_k(const float* __restrict__ x, float* __restrict__ o, int n) {
  __shared__ float sm[256];
  int b = blockIdx.x;
  float m = -3.4e38f;
  for (int j = threadIdx.x; j < n; j += 256) m = fmaxf(m, x[(long long)b * n + j]);
  sm[threadIdx.x] = m;
  __syncthreads();
  for (int s = 128; s > 0; s >>= 1) {
    if ((int)threadIdx.x < s) sm[threadIdx.x] = fmaxf(sm[threadIdx.x], sm[threadIdx.x + s]);
    __syncthreads();
  }
  if (threadIdx.x == 0) o[b] = sm[0];
}
__global__ void exp_sub_k(const float* x, const float* mx, float* o, long long n, int Fdim) {
  long long i = (long long)blockIdx.x * blockDim.x + threadIdx.x;
  if (i < n) o[i] = expf(x[i] - mx[i / Fdim]);
}
__global__ void scatter_add1_k(const float* v, const int* __restrict__ idx, float* o,
                               long long n, int Fdim, int Edim) {
  long long i = (long long)blockIdx.x * blockDim.x + threadIdx.x;
  if (i >= n) return;
  long long b = i / Fdim;
  atomicAdd(o + b * Edim + idx[i], v[i]);
}
__global__ void clamp_min_k(float* p, float v, long long n) {
  long long i = (long long)blockIdx.x * blockDim.x + threadIdx.x;
  if (i < n) p[i] = fmaxf(p[i], v);
}
__global__ void norm_k(const float* Wtil, const float* pr, const float* e2fs,
                       const int* __restrict__ head, float* o, long long n) {
  long long i = (long long)blockIdx.x * blockDim.x + threadIdx.x;
  if (i >= n) return;
  long long b = i / cf::F;
  int h = head[i];
  o[i] = Wtil[i] * pr[b * cf::E + h] / e2fs[b * cf::E + h];
}
// e2f_emb (f16, ld=128, pad=0) = relu(kbs_f + kbh_e[b,head,t]) * norm[b,f]
__global__ void e2f_ew_k(const float* __restrict__ kbsf, const float* __restrict__ kbh,
                         const float* __restrict__ norm, const int* __restrict__ head,
                         _Float16* __restrict__ o, long long n /* B*F*128 */) {
  long long i = (long long)blockIdx.x * blockDim.x + threadIdx.x;
  if (i >= n) return;
  long long be = i >> 7;               // (b*F+f)
  int t = (int)(i & 127);
  if (t >= cf::T) { o[i] = (_Float16)0.0f; return; }
  long long b = be / cf::F;
  int h = head[be];
  float v = fmaxf(kbsf[be * cf::T + t] + kbh[((long long)b * cf::E + h) * cf::T + t], 0.0f);
  o[i] = (_Float16)(v * norm[be]);
}
__global__ void scatter_vec_k(const float* v, const int* __restrict__ tail, float* o,
                              long long n) {
  long long i = (long long)blockIdx.x * blockDim.x + threadIdx.x;
  if (i >= n) return;
  long long b = i / ((long long)cf::F * cf::T);
  long long rem = i - b * ((long long)cf::F * cf::T);
  int f = (int)(rem / cf::T);
  int t = (int)(rem - (long long)f * cf::T);
  int e = tail[b * cf::F + f];
  atomicAdd(o + ((long long)b * cf::E + e) * cf::T + t, v[i]);
}
__global__ void relu_add_k(float* a, const float* b, long long n) {
  long long i = (long long)blockIdx.x * blockDim.x + threadIdx.x;
  if (i < n) a[i] = fmaxf(a[i] + b[i], 0.0f);
}
__global__ void pr_update_k(float* pr, const float* sc, long long n) {
  long long i = (long long)blockIdx.x * blockDim.x + threadIdx.x;
  if (i < n) pr[i] = cf::LAMBDA * sc[i] + (1.0f - cf::LAMBDA) * pr[i];
}
// next_emb f16 (B,E,320): [ent | qn2(b) | f2e | 0]
__global__ void next_build_k(const float* __restrict__ ent, const float* __restrict__ qn2,
                             const float* __restrict__ f2e, _Float16* __restrict__ nxt,
                             long long n) {
  long long i = (long long)blockIdx.x * blockDim.x + threadIdx.x;
  if (i >= n) return;
  long long be = i / 320;
  int j = (int)(i - be * 320);
  long long b = be / cf::E;
  float v;
  if (j < 100)      v = ent[be * cf::T + j];
  else if (j < 200) v = qn2[b * cf::T + (j - 100)];
  else if (j < 300) v = f2e[be * cf::T + (j - 200)];
  else              v = 0.0f;
  nxt[i] = (_Float16)v;
}
// s16[b,:320] = f16(sum_e pr[b,e]*[ent|qn2|f2e|0]);  prsum[b] = sum_e pr[b,e]
__global__ void svec_k(const float* __restrict__ pr, const float* __restrict__ ent,
                       const float* __restrict__ qn2, const float* __restrict__ f2e,
                       _Float16* __restrict__ s16, float* __restrict__ prsum) {
  int b = blockIdx.x;
  int j = threadIdx.x;  // 0..319
  const float* prb = pr + (long long)b * cf::E;
  float acc = 0.0f;
  if (j < 100) {
    for (int e = 0; e < cf::E; ++e) acc += prb[e] * ent[((long long)b * cf::E + e) * cf::T + j];
  } else if (j < 200) {
    float ps = 0.0f;
    for (int e = 0; e < cf::E; ++e) ps += prb[e];
    acc = ps * qn2[b * cf::T + (j - 100)];
    if (j == 100) prsum[b] = ps;
  } else if (j < 300) {
    for (int e = 0; e < cf::E; ++e)
      acc += prb[e] * f2e[((long long)b * cf::E + e) * cf::T + (j - 200)];
  }
  s16[(long long)b * 320 + j] = (_Float16)acc;
}

// ---------------------------------------------------------------------------
extern "C" void kernel_launch(void* const* d_in, const int* in_sizes, int n_in,
                              void* d_out, int out_size, void* d_ws, size_t ws_size,
                              hipStream_t stream) {
  (void)in_sizes; (void)n_in; (void)out_size; (void)ws_size;
  using namespace cf;
  const float* word_emb   = (const float*)d_in[0];
  const float* entity_emb = (const float*)d_in[1];
  const float* ent_W      = (const float*)d_in[2];
  const float* ent_b      = (const float*)d_in[3];
  const float* lstm_Wih   = (const float*)d_in[4];
  const float* lstm_Whh   = (const float*)d_in[5];
  const float* lstm_bih   = (const float*)d_in[6];
  const float* lstm_bhh   = (const float*)d_in[7];
  const float* q2e_W      = (const float*)d_in[8];
  const float* q2e_b      = (const float*)d_in[9];
  const float* e2q_W      = (const float*)d_in[10];
  const float* e2q_b      = (const float*)d_in[11];
  const float* e2e_W      = (const float*)d_in[12];
  const float* e2e_b      = (const float*)d_in[13];
  const float* kbh_W      = (const float*)d_in[14];
  const float* kbh_b      = (const float*)d_in[15];
  const float* kbt_W      = (const float*)d_in[16];
  const float* kbt_b      = (const float*)d_in[17];
  const float* kbs_W      = (const float*)d_in[18];
  const float* kbs_b      = (const float*)d_in[19];
  const float* q2e_adj    = (const float*)d_in[20];
  const int*   query_text = (const int*)d_in[21];
  const int*   local_ent  = (const int*)d_in[22];
  const int*   kb_fact    = (const int*)d_in[23];
  const int*   head_idx   = (const int*)d_in[24];
  const int*   tail_idx   = (const int*)d_in[25];

  float* ws = (float*)d_ws;
  size_t off = 0;
  auto alloc = [&](size_t nf) { float* p = ws + off; off += (nf + 63) & ~(size_t)63; return p; };
  auto alloch = [&](size_t nh) { return (_Float16*)alloc((nh + 1) / 2); };

  // fp32 buffers
  float* bias_lstm = alloc(400);
  float* xw    = alloc((size_t)B * LQ * 400);
  float* qnode = alloc((size_t)B * T);
  float* fact  = alloc((size_t)B * F * T);
  float* ent   = alloc((size_t)B * E * T);
  float* bufF1 = alloc((size_t)B * F * T);       // att -> kbs_f -> kbt_f
  float* simTf = alloc((size_t)B * F * 32);
  float* Wsc   = alloc((size_t)B * F);
  float* Wtil  = alloc((size_t)B * F);
  float* rmax  = alloc(B);
  float* e2fs  = alloc((size_t)B * E);
  float* pr    = alloc((size_t)B * E);
  float* prsc  = alloc((size_t)B * E);
  float* qn2   = alloc((size_t)B * T);
  float* kbh_e = alloc((size_t)B * E * T);
  float* scat  = alloc((size_t)B * E * T);
  float* kbse  = alloc((size_t)B * E * T);       // becomes f2e in place
  float* prsum = alloc(B);
  float* norm  = alloc((size_t)B * F);
  // f16 buffers (padded leading dims, multiples of 32 halfs)
  _Float16* embH   = alloch((size_t)50007 * 128);  // entity table, EV x 128
  _Float16* entW16 = alloch((size_t)T * 128);
  _Float16* Wih16  = alloch((size_t)400 * 320);
  _Float16* Whh16  = alloch((size_t)400 * 128);
  _Float16* q2eW16 = alloch((size_t)2 * T * 128);
  _Float16* kbhW16 = alloch((size_t)2 * T * 128);
  _Float16* kbtW16 = alloch((size_t)2 * T * 128);
  _Float16* kbsW16 = alloch((size_t)2 * T * 128);
  _Float16* e2qW16 = alloch((size_t)2 * T * 320);
  _Float16* e2eW16 = alloch((size_t)2 * T * 320);
  _Float16* Xg16   = alloch((size_t)B * LQ * 320);
  _Float16* Qh16   = alloch((size_t)B * LQ * 128);
  _Float16* QT16   = alloch((size_t)B * T * 32);
  _Float16* qnd16  = alloch((size_t)B * 128);
  _Float16* fact16 = alloch((size_t)B * F * 128);
  _Float16* ent16  = alloch((size_t)B * E * 128);
  _Float16* simT16 = alloch((size_t)B * F * 32);
  _Float16* bufF2h = alloch((size_t)B * F * 128); // e2f_emb ; also reused as nxt16
  _Float16* s16    = alloch((size_t)B * 320);
  _Float16* nxt16  = bufF2h;                      // (B*E*320 <= B*F*128)

  auto gemm = [&](const _Float16* X, const _Float16* W, const float* bias,
                  const float* brs, const int* idx, float* Y, _Float16* Yh,
                  int M, int N, int Kp, int lda, int ldb, int ldy, int ldyh,
                  long long xbs, long long wbs, long long ybs, long long ybsh,
                  float alpha, int flags, int batches) {
    dim3 grid((M + 63) / 64, (N + 31) / 32, batches);
    wmma_gemm<<<grid, dim3(256), 0, stream>>>(X, W, bias, brs, idx, Y, Yh, M, N, Kp,
                                              lda, ldb, ldy, ldyh, xbs, wbs, ybs, ybsh,
                                              alpha, flags);
  };
  auto cvt = [&](const float* s, _Float16* d, long long rows, int K, int ldp) {
    cvt_pad_k<<<(unsigned)((rows * ldp + 255) / 256), 256, 0, stream>>>(s, d, rows, K, K, ldp);
  };
  auto fill = [&](float* p, float v, long long n) {
    fill_k<<<(unsigned)((n + 255) / 256), 256, 0, stream>>>(p, v, n);
  };

  const float invdiv = 0.1f;  // 1/sqrt(T)

  // ---- one-time f16 conversions (weights + entity table + gathered words) ----
  cvt(entity_emb, embH, 50007, T, 128);
  cvt(ent_W, entW16, T, T, 128);
  cvt(lstm_Wih, Wih16, 400, EMB, 320);
  cvt(lstm_Whh, Whh16, 400, T, 128);
  cvt(q2e_W, q2eW16, 2 * T, T, 128);
  cvt(kbh_W, kbhW16, 2 * T, T, 128);
  cvt(kbt_W, kbtW16, 2 * T, T, 128);
  cvt(kbs_W, kbsW16, 2 * T, T, 128);
  cvt(e2q_W, e2qW16, 2 * T, K3, 320);
  cvt(e2e_W, e2eW16, 2 * T, K3, 320);
  gather_cvt_k<<<(B * LQ * 320 + 255) / 256, 256, 0, stream>>>(
      word_emb, query_text, Xg16, B * LQ, EMB, EMB, 320);

  // 1) xw = gathered words @ Wih^T + (bih+bhh)
  vec_add_k<<<2, 256, 0, stream>>>(lstm_bih, lstm_bhh, bias_lstm, 400);
  gemm(Xg16, Wih16, bias_lstm, nullptr, nullptr, xw, nullptr,
       B * LQ, 400, 320, 320, 320, 400, 0, 0, 0, 0, 0, 1.0f, 0, 1);

  // 2) LSTM recurrence -> Qh16, QT16, qnd16
  lstm_kernel<<<1, 1024, 64 * 128 * 2 + 64 * 400 * 4 + 64 * 100 * 4, stream>>>(
      xw, Whh16, Qh16, QT16, qnd16);

  // 3) gather + linear over entity table (fp32 + f16 shadow outputs)
  gemm(embH, entW16, ent_b, nullptr, kb_fact, fact, fact16,
       B * F, T, 128, 128, 128, T, 128, 0, 0, 0, 0, 1.0f, GF_STOREH, 1);
  gemm(embH, entW16, ent_b, nullptr, local_ent, ent, ent16,
       B * E, T, 128, 128, 128, T, 128, 0, 0, 0, 0, 1.0f, GF_STOREH, 1);

  // 4) sim^T = (Q @ fact^T / div)^T, masked softmax over L -> f16
  gemm(Qh16, fact16, nullptr, nullptr, nullptr, simTf, nullptr,
       LQ, F, 128, 128, 128, 32, 0, (long long)LQ * 128, (long long)F * 128,
       (long long)F * 32, 0, invdiv, GF_TRANS, B);
  softmax_l_k<<<(B * F + 255) / 256, 256, 0, stream>>>(simTf, simT16, query_text, B * F, F);

  // 5) fact2query_att = simT @ (Q^T)^T (K=32), then Wsc, W_tilde, e2f_softmax
  gemm(simT16, QT16, nullptr, nullptr, nullptr, bufF1, nullptr,
       F, T, 32, 32, 32, T, 0, (long long)F * 32, (long long)T * 32,
       (long long)F * T, 0, 1.0f, 0, B);
  rowdot_k<<<(B * F + 7) / 8, 256, 0, stream>>>(bufF1, fact, Wsc, (long long)B * F, T, invdiv);
  rowmax_k<<<B, 256, 0, stream>>>(Wsc, rmax, F);
  exp_sub_k<<<(B * F + 255) / 256, 256, 0, stream>>>(Wsc, rmax, Wtil, (long long)B * F, F);
  fill(e2fs, 0.0f, (long long)B * E);
  scatter_add1_k<<<(B * F + 255) / 256, 256, 0, stream>>>(Wtil, head_idx, e2fs,
                                                          (long long)B * F, F, E);
  clamp_min_k<<<(B * E + 255) / 256, 256, 0, stream>>>(e2fs, EPSV, (long long)B * E);
  hipMemcpyAsync(pr, q2e_adj, (size_t)B * E * sizeof(float), hipMemcpyDeviceToDevice, stream);

  const long long nBFT = (long long)B * F * T;
  const long long nBET = (long long)B * E * T;

  for (int i = 0; i < 2; ++i) {
    const _Float16* q2eWi = q2eW16 + (size_t)i * T * 128;
    const _Float16* kbhWi = kbhW16 + (size_t)i * T * 128;
    const _Float16* kbtWi = kbtW16 + (size_t)i * T * 128;
    const _Float16* kbsWi = kbsW16 + (size_t)i * T * 128;
    const _Float16* e2qWi = e2qW16 + (size_t)i * T * 320;
    const _Float16* e2eWi = e2eW16 + (size_t)i * T * 320;
    const float* q2ebi = q2e_b + (size_t)i * T;
    const float* kbhbi = kbh_b + (size_t)i * T;
    const float* kbtbi = kbt_b + (size_t)i * T;
    const float* kbsbi = kbs_b + (size_t)i * T;
    const float* e2qbi = e2q_b + (size_t)i * T;
    const float* e2ebi = e2e_b + (size_t)i * T;

    // q2e component (depends only on b)
    gemm(qnd16, q2eWi, q2ebi, nullptr, nullptr, qn2, nullptr,
         B, T, 128, 128, 128, T, 0, 0, 0, 0, 0, 1.0f, 0, 1);
    // kbh on entities, kbs on facts
    gemm(ent16, kbhWi, kbhbi, nullptr, nullptr, kbh_e, nullptr,
         B * E, T, 128, 128, 128, T, 0, 0, 0, 0, 0, 1.0f, 0, 1);
    gemm(fact16, kbsWi, kbsbi, nullptr, nullptr, bufF1, nullptr,
         B * F, T, 128, 128, 128, T, 0, 0, 0, 0, 0, 1.0f, 0, 1);
    // e2f_norm, gated fact messages (f16 out, padded)
    norm_k<<<(B * F + 255) / 256, 256, 0, stream>>>(Wtil, pr, e2fs, head_idx, norm,
                                                    (long long)B * F);
    e2f_ew_k<<<(unsigned)(((long long)B * F * 128 + 255) / 256), 256, 0, stream>>>(
        bufF1, kbh_e, norm, head_idx, bufF2h, (long long)B * F * 128);
    // kbt on fact messages, scatter to tail entities
    gemm(bufF2h, kbtWi, kbtbi, nullptr, nullptr, bufF1, nullptr,
         B * F, T, 128, 128, 128, T, 0, 0, 0, 0, 0, 1.0f, 0, 1);
    fill(scat, 0.0f, nBET);
    scatter_vec_k<<<(unsigned)((nBFT + 255) / 256), 256, 0, stream>>>(bufF1, tail_idx,
                                                                      scat, nBFT);
    // f2e = relu(kbs(entity) + scatter), in place in kbse
    gemm(ent16, kbsWi, kbsbi, nullptr, nullptr, kbse, nullptr,
         B * E, T, 128, 128, 128, T, 0, 0, 0, 0, 0, 1.0f, 0, 1);
    relu_add_k<<<(unsigned)((nBET + 255) / 256), 256, 0, stream>>>(kbse, scat, nBET);
    // pagerank update
    fill(prsc, 0.0f, (long long)B * E);
    scatter_add1_k<<<(B * F + 255) / 256, 256, 0, stream>>>(norm, tail_idx, prsc,
                                                            (long long)B * F, F, E);
    pr_update_k<<<(B * E + 255) / 256, 256, 0, stream>>>(pr, prsc, (long long)B * E);
    // e2q folded through linearity: qnode = (sum_e pr*next) @ W^T + (sum_e pr)*b
    svec_k<<<B, 320, 0, stream>>>(pr, ent, qn2, kbse, s16, prsum);
    gemm(s16, e2qWi, e2qbi, prsum, nullptr, qnode, qnd16,
         B, T, 320, 320, 320, T, 128, 0, 0, 0, 0, 1.0f, GF_STOREH, 1);
    // next_emb (f16, B*E*320) then local_entity_emb = relu(next @ e2e^T + b)
    next_build_k<<<(unsigned)(((long long)B * E * 320 + 255) / 256), 256, 0, stream>>>(
        ent, qn2, kbse, nxt16, (long long)B * E * 320);
    gemm(nxt16, e2eWi, e2ebi, nullptr, nullptr, ent, ent16,
         B * E, T, 320, 320, 320, T, 128, 0, 0, 0, 0, 1.0f, GF_RELU | GF_STOREH, 1);
  }

  // outputs: [local_entity_emb (B*E*T) | query_node_emb (B*T) | pagerank (B*E)]
  float* out = (float*)d_out;
  hipMemcpyAsync(out, ent, (size_t)B * E * T * sizeof(float), hipMemcpyDeviceToDevice, stream);
  hipMemcpyAsync(out + (size_t)B * E * T, qnode, (size_t)B * T * sizeof(float),
                 hipMemcpyDeviceToDevice, stream);
  hipMemcpyAsync(out + (size_t)B * E * T + (size_t)B * T, pr, (size_t)B * E * sizeof(float),
                 hipMemcpyDeviceToDevice, stream);
}
